// fVSRN_6141803233745
// MI455X (gfx1250) — compile-verified, matrix operand order checked by reading, not run on hardware
//
#include <hip/hip_runtime.h>
#include <hip/hip_bf16.h>

typedef __attribute__((ext_vector_type(16))) _Float16 v16h;
typedef __attribute__((ext_vector_type(8)))  float    v8f;

#define N_PTS      1048576
#define GRID_R     96
#define YS         96
#define ZS         (96*96)
#define CS         (96*96*96)     // channel stride = 884736
#define D_IN_PAD   96             // 68 padded to 3 K-steps of 32
#define HD         128
#define W0H_OFF    0              // 128*96  halves
#define W1H_OFF    (128*96)       // 128*128 halves
#define W2H_OFF    (128*96 + 128*128)  // 16*128 halves
#define WS_HALVES  (128*96 + 128*128 + 16*128)

// Load a 16-half WMMA fragment: elements 0..7 at p[0..7], 8..15 at p[16..23].
// p must be 16-byte aligned.
__device__ __forceinline__ v16h load_frag(const _Float16* p) {
  union { v16h v; float4 q[2]; } u;
  u.q[0] = *reinterpret_cast<const float4*>(p);
  u.q[1] = *reinterpret_cast<const float4*>(p + 16);
  return u.v;
}

__device__ __forceinline__ v8f wmma_f16(v16h a, v16h b, v8f c) {
  return __builtin_amdgcn_wmma_f32_16x16x32_f16(false, a, false, b, (short)0, c,
                                                false, false);
}

// ---------------- prep: f32 weights -> padded f16 weights in workspace ------
__global__ __launch_bounds__(256) void fvsrn_prep(
    const float* __restrict__ W0,   // [128][68]
    const float* __restrict__ W1,   // [128][128]
    const float* __restrict__ W2,   // [1][128]
    _Float16* __restrict__ ws) {
  int i = blockIdx.x * 256 + threadIdx.x;
  if (i < 128 * D_IN_PAD) {                       // W0h [128][96], K-padded
    int j = i / D_IN_PAD, k = i % D_IN_PAD;
    ws[W0H_OFF + i] = (k < 68) ? (_Float16)W0[j * 68 + k] : (_Float16)0.f;
    return;
  }
  i -= 128 * D_IN_PAD;
  if (i < 128 * 128) {                            // W1h [128][128]
    ws[W1H_OFF + i] = (_Float16)W1[i];
    return;
  }
  i -= 128 * 128;
  if (i < 16 * 128) {                             // W2h [16][128], N-padded
    int j = i / 128, k = i % 128;
    ws[W2H_OFF + i] = (j == 0) ? (_Float16)W2[k] : (_Float16)0.f;
  }
}

// ---------------- main: features + 3-layer MLP via WMMA ---------------------
__global__ __launch_bounds__(256) void fvsrn_main(
    const float* __restrict__ x,        // [N][3]
    const float* __restrict__ grid,     // [32][96][96][96]
    const _Float16* __restrict__ W0h,   // [128][96]
    const float* __restrict__ b0,       // [128]
    const _Float16* __restrict__ W1h,   // [128][128]
    const float* __restrict__ b1,       // [128]
    const _Float16* __restrict__ W2h,   // [16][128]
    const float* __restrict__ b2,       // [1]
    float* __restrict__ out) {          // [N]
  __shared__ __align__(16) _Float16 bufA[128 * D_IN_PAD];  // layer-0 input
  __shared__ __align__(16) _Float16 bufB[128 * HD];        // hidden acts

  const int tid  = threadIdx.x;
  const int lane = tid & 31;
  const int wave = tid >> 5;          // 0..7 -> owns M-band rows [16w,16w+16)
  const int p2   = tid >> 1;          // 0..127  local point
  const int hf   = tid & 1;           // half-point worker
  const long pt  = (long)blockIdx.x * 128 + p2;

  // ---------------- phase 1: PE + trilinear gather into bufA --------------
  {
    const float px = x[pt * 3 + 0];
    const float py = x[pt * 3 + 1];
    const float pz = x[pt * 3 + 2];
    _Float16* row = bufA + p2 * D_IN_PAD;

    // positional encoding: half hf does freqs i = 3*hf .. 3*hf+2
#pragma unroll
    for (int i = 0; i < 3; ++i) {
      const int fi = hf * 3 + i;
      const float f = 3.14159265358979323846f * (float)(1 << fi);
      row[6 * fi + 0] = (_Float16)__sinf(px * f);
      row[6 * fi + 1] = (_Float16)__sinf(py * f);
      row[6 * fi + 2] = (_Float16)__sinf(pz * f);
      row[6 * fi + 3] = (_Float16)__cosf(px * f);
      row[6 * fi + 4] = (_Float16)__cosf(py * f);
      row[6 * fi + 5] = (_Float16)__cosf(pz * f);
    }

    // trilinear setup (align_corners=True; grid_sample coord order x->W,y->H,z->D)
    const float fx = (px + 1.f) * 0.5f * (GRID_R - 1);
    const float fy = (py + 1.f) * 0.5f * (GRID_R - 1);
    const float fz = (pz + 1.f) * 0.5f * (GRID_R - 1);
    const float x0f = floorf(fx), y0f = floorf(fy), z0f = floorf(fz);
    const float wx = fx - x0f, wy = fy - y0f, wz = fz - z0f;
    auto clampi = [](int v) { return v < 0 ? 0 : (v > GRID_R - 1 ? GRID_R - 1 : v); };
    const int x0 = clampi((int)x0f), x1 = clampi((int)x0f + 1);
    const int y0 = clampi((int)y0f), y1 = clampi((int)y0f + 1);
    const int z0 = clampi((int)z0f), z1 = clampi((int)z0f + 1);
    const float wx0 = 1.f - wx, wy0 = 1.f - wy, wz0 = 1.f - wz;
    const float w000 = wz0 * wy0 * wx0, w001 = wz0 * wy0 * wx;
    const float w010 = wz0 * wy  * wx0, w011 = wz0 * wy  * wx;
    const float w100 = wz  * wy0 * wx0, w101 = wz  * wy0 * wx;
    const float w110 = wz  * wy  * wx0, w111 = wz  * wy  * wx;
    const int o000 = z0 * ZS + y0 * YS + x0, o001 = z0 * ZS + y0 * YS + x1;
    const int o010 = z0 * ZS + y1 * YS + x0, o011 = z0 * ZS + y1 * YS + x1;
    const int o100 = z1 * ZS + y0 * YS + x0, o101 = z1 * ZS + y0 * YS + x1;
    const int o110 = z1 * ZS + y1 * YS + x0, o111 = z1 * ZS + y1 * YS + x1;

    // half hf gathers channels hf*16 .. hf*16+15 (grid is L2-resident: 113MB < 192MB)
    for (int c = 0; c < 16; ++c) {
      const int cc = hf * 16 + c;
      const float* g = grid + (size_t)cc * CS;
      float v = w000 * g[o000] + w001 * g[o001] + w010 * g[o010] + w011 * g[o011]
              + w100 * g[o100] + w101 * g[o101] + w110 * g[o110] + w111 * g[o111];
      row[36 + cc] = (_Float16)v;
    }
    // zero K-padding 68..95 (14 each half)
#pragma unroll
    for (int j = 0; j < 14; ++j) row[68 + hf * 14 + j] = (_Float16)0.f;
  }
  __syncthreads();

  const int mlane = lane & 15;
  const int sel   = lane >> 4;             // K sub-group select per ISA layout
  const int mbase = wave * 16 + sel * 8;   // rows this lane's C fragment covers

  // ---------------- layer 0: [128x96] @ W0h^T -> ReLU -> bufB -------------
  {
    const _Float16* arow = bufA + (wave * 16 + mlane) * D_IN_PAD + sel * 8;
    const v16h a0 = load_frag(arow + 0 * 32);
    const v16h a1 = load_frag(arow + 1 * 32);
    const v16h a2 = load_frag(arow + 2 * 32);
#pragma unroll
    for (int nt = 0; nt < 8; ++nt) {
      const int j = nt * 16 + mlane;       // output neuron = B column
      const _Float16* brow = W0h + j * D_IN_PAD + sel * 8;
      v8f acc = {};
      acc = wmma_f16(a0, load_frag(brow + 0 * 32), acc);
      acc = wmma_f16(a1, load_frag(brow + 1 * 32), acc);
      acc = wmma_f16(a2, load_frag(brow + 2 * 32), acc);
      const float bias = b0[j];
#pragma unroll
      for (int r = 0; r < 8; ++r) {
        float v = acc[r] + bias;
        v = v > 0.f ? v : 0.f;
        bufB[(mbase + r) * HD + j] = (_Float16)v;
      }
    }
  }
  __syncthreads();

  // ---------------- layer 1: [128x128] @ W1h^T -> ReLU, in-place ----------
  {
    const _Float16* arow = bufB + (wave * 16 + mlane) * HD + sel * 8;
    v16h a[4];
#pragma unroll
    for (int kk = 0; kk < 4; ++kk) a[kk] = load_frag(arow + kk * 32);
#pragma unroll
    for (int nt = 0; nt < 8; ++nt) {
      const int j = nt * 16 + mlane;
      const _Float16* brow = W1h + j * HD + sel * 8;
      v8f acc = {};
#pragma unroll
      for (int kk = 0; kk < 4; ++kk)
        acc = wmma_f16(a[kk], load_frag(brow + kk * 32), acc);
      const float bias = b1[j];
#pragma unroll
      for (int r = 0; r < 8; ++r) {
        float v = acc[r] + bias;
        v = v > 0.f ? v : 0.f;
        bufB[(mbase + r) * HD + j] = (_Float16)v;   // band-local in-place
      }
    }
  }
  __syncthreads();

  // ---------------- layer 2: [128x128] @ W2h^T (N padded to 16) -----------
  {
    const _Float16* arow = bufB + (wave * 16 + mlane) * HD + sel * 8;
    const _Float16* brow = W2h + mlane * HD + sel * 8;   // row j = mlane (only j=0 real)
    v8f acc = {};
#pragma unroll
    for (int kk = 0; kk < 4; ++kk)
      acc = wmma_f16(load_frag(arow + kk * 32), load_frag(brow + kk * 32), acc);
    if (mlane == 0) {                       // column N=0 holds the real output
      const float bv = b2[0];
      const long obase = (long)blockIdx.x * 128 + mbase;
#pragma unroll
      for (int r = 0; r < 8; ++r)
        out[obase + r] = (acc[r] + bv) * 2.f - 1.f;   // *(VMAX-VMIN)+VMIN
    }
  }
}

extern "C" void kernel_launch(void* const* d_in, const int* in_sizes, int n_in,
                              void* d_out, int out_size, void* d_ws, size_t ws_size,
                              hipStream_t stream) {
  const float* x    = (const float*)d_in[0];
  const float* grid = (const float*)d_in[1];
  const float* W0   = (const float*)d_in[2];
  const float* b0   = (const float*)d_in[3];
  const float* W1   = (const float*)d_in[4];
  const float* b1   = (const float*)d_in[5];
  const float* W2   = (const float*)d_in[6];
  const float* b2   = (const float*)d_in[7];
  _Float16* wsh = (_Float16*)d_ws;

  fvsrn_prep<<<(WS_HALVES + 255) / 256, 256, 0, stream>>>(W0, W1, W2, wsh);
  fvsrn_main<<<N_PTS / 128, 256, 0, stream>>>(
      x, grid, wsh + W0H_OFF, b0, wsh + W1H_OFF, b1, wsh + W2H_OFF, b2,
      (float*)d_out);
}
